// GATNet_12695923327560
// MI455X (gfx1250) — compile-verified, matrix operand order checked by reading, not run on hardware
//
#include <hip/hip_runtime.h>
#include <hip/hip_bf16.h>

// ---------------------------------------------------------------------------
// CDNA5 (gfx1250, wave32) GAT + TransE pipeline.
//   GEMM:   v_wmma_f32_16x16x32_bf16, f32 accumulate.
//   Edges:  recompute-based 3-pass segment softmax with u32-ordered float max
//           and native f32 atomic adds (memory-bound part of the workload).
// ---------------------------------------------------------------------------

typedef __attribute__((ext_vector_type(16))) __bf16 v16bf;
typedef __attribute__((ext_vector_type(8)))  float  v8f;

#define DIMK   128
#define HEADS  8
#define DHEAD  16
#define NEG_SLOPE 0.2f

// f32 -> bf16: use the native conversion (RNE fptrunc) so the backend can emit
// v_cvt_pk_bf16_f32 instead of a 3-op bit-twiddle sequence per element.
static __device__ __forceinline__ __bf16 f2bf(float f) {
    return (__bf16)f;
}

// Monotone order-preserving float<->uint mapping for atomicMax on floats.
static __device__ __forceinline__ unsigned f2ord(float f) {
    unsigned u = __float_as_uint(f);
    return (u & 0x80000000u) ? ~u : (u | 0x80000000u);
}
static __device__ __forceinline__ float ord2f(unsigned u) {
    unsigned b = (u & 0x80000000u) ? (u ^ 0x80000000u) : ~u;
    return __uint_as_float(b);
}

// ---------------------------------------------------------------------------
// Pack W[layer] (f32 [H][128][16]) into the wave32 WMMA B-fragment layout:
// gB[((ks*8 + ct)*32 + lane)*16 + i]  (bf16), ks = k-step (K0 = ks*32),
// ct = 16-col tile (== head), lane/i per ISA 16-bit A-layout with N<->M.
//   i <  8 : K = ks*32 + (lane>>4)*8 + i
//   i >= 8 : K = ks*32 + 16 + (lane>>4)*8 + (i-8)
//   N      = ct*16 + (lane&15)       -> B[K][N] = W[l][ct][K][lane&15]
// ---------------------------------------------------------------------------
__global__ void prep_B_kernel(const float* __restrict__ W, int layer,
                              __bf16* __restrict__ gB) {
    for (int e = threadIdx.x; e < 4 * 8 * 32 * 16; e += 256) {
        int i    = e & 15;
        int lane = (e >> 4) & 31;
        int ct   = (e >> 9) & 7;
        int ks   = (e >> 12) & 3;
        int kk   = lane & 15;
        int k    = ks * 32 + ((lane >> 4) << 3) + (i & 7) + ((i >> 3) << 4);
        float v  = W[(((size_t)layer * HEADS + ct) * DIMK + k) * DHEAD + kk];
        gB[e] = f2bf(v);
    }
}

// ---------------------------------------------------------------------------
// Wh[N,128] = h[N,128] @ B[128,128] via WMMA. One wave -> 16 rows x 128 cols.
// 8 waves per block -> 128 rows per block. B fragments staged in 32KB LDS.
// ---------------------------------------------------------------------------
__global__ void __launch_bounds__(256)
gat_gemm_wmma(const float* __restrict__ h, const __bf16* __restrict__ gB,
              float* __restrict__ Wh, int N) {
    __shared__ v16bf sB[4][8][32];                 // 32 KB
    {
        const float4* s = (const float4*)gB;
        float4*       d = (float4*)sB;
        #pragma unroll
        for (int i = 0; i < 8; ++i) d[threadIdx.x + i * 256] = s[threadIdx.x + i * 256];
    }
    __syncthreads();

    const int wave    = threadIdx.x >> 5;
    const int lane    = threadIdx.x & 31;
    const int rowBase = blockIdx.x * 128 + wave * 16;
    if (rowBase >= N) return;                      // wave-uniform exit: EXEC stays full

    const int half   = lane >> 4;
    int       mrow   = rowBase + (lane & 15);
    if (mrow >= N) mrow = N - 1;                   // clamp loads; stores masked later
    const float* hrow = h + (size_t)mrow * DIMK;

    v8f acc[8] = {};

    #pragma unroll
    for (int ks = 0; ks < 4; ++ks) {
        const int k0 = ks * 32 + half * 8;
        v16bf a;
        #pragma unroll
        for (int i = 0; i < 8; ++i) {
            a[i]     = f2bf(hrow[k0 + i]);         // K = k0..k0+7
            a[i + 8] = f2bf(hrow[k0 + 16 + i]);    // K = k0+16..k0+23
        }
        #pragma unroll
        for (int ct = 0; ct < 8; ++ct) {
            v16bf b = sB[ks][ct][lane];
            acc[ct] = __builtin_amdgcn_wmma_f32_16x16x32_bf16(
                false, a, false, b, (short)0, acc[ct], false, false);
        }
    }

    // D layout: lane holds col n = ct*16 + (lane&15); vgpr r -> row half*8 + r.
    const int n = lane & 15;
    #pragma unroll
    for (int ct = 0; ct < 8; ++ct) {
        #pragma unroll
        for (int r = 0; r < 8; ++r) {
            int m = rowBase + half * 8 + r;
            if (m < N) Wh[(size_t)m * DIMK + ct * 16 + n] = acc[ct][r];
        }
    }
}

// es[n,h] = Wh[n,h,:].a_src[h,:],  ed likewise. One thread per (n,h).
__global__ void attn_coef_kernel(const float* __restrict__ Wh,
                                 const float* __restrict__ a_src,
                                 const float* __restrict__ a_dst,
                                 float* __restrict__ es, float* __restrict__ ed,
                                 int N) {
    int t = blockIdx.x * blockDim.x + threadIdx.x;
    if (t >= N * HEADS) return;
    int n = t >> 3, hh = t & 7;
    const float* w  = Wh + (size_t)n * DIMK + hh * DHEAD;
    const float* as = a_src + hh * DHEAD;
    const float* ad = a_dst + hh * DHEAD;
    float s = 0.f, d = 0.f;
    #pragma unroll
    for (int k = 0; k < DHEAD; ++k) { float x = w[k]; s += x * as[k]; d += x * ad[k]; }
    es[t] = s; ed[t] = d;
}

__global__ void zero_kernel(float* __restrict__ acc, long long n1,
                            unsigned* __restrict__ mu, float* __restrict__ denom,
                            long long n2) {
    long long t = (long long)blockIdx.x * blockDim.x + threadIdx.x;
    if (t < n1) acc[t] = 0.f;
    if (t < n2) { mu[t] = 0u; denom[t] = 0.f; }   // 0u == ordered(-NaN) == global min
}

// Pass 1: segment max of leaky_relu(es[src]+ed[dst]) into mu[dst] (ordered u32).
__global__ void edge_max_kernel(const int* __restrict__ src, const int* __restrict__ dst,
                                int E, int N,
                                const float* __restrict__ es, const float* __restrict__ ed,
                                unsigned* __restrict__ mu) {
    int t = blockIdx.x * blockDim.x + threadIdx.x;
    if (t >= E + N) return;
    int s = t < E ? src[t] : t - E;                // virtual self-loop edges
    int d = t < E ? dst[t] : t - E;
    #pragma unroll
    for (int h = 0; h < HEADS; ++h) {
        float e = es[s * HEADS + h] + ed[d * HEADS + h];
        e = e > 0.f ? e : NEG_SLOPE * e;
        atomicMax(&mu[d * HEADS + h], f2ord(e));
    }
}

// Pass 2: denom[dst] += exp(e - m[dst]).
__global__ void edge_sum_kernel(const int* __restrict__ src, const int* __restrict__ dst,
                                int E, int N,
                                const float* __restrict__ es, const float* __restrict__ ed,
                                const unsigned* __restrict__ mu,
                                float* __restrict__ denom) {
    int t = blockIdx.x * blockDim.x + threadIdx.x;
    if (t >= E + N) return;
    int s = t < E ? src[t] : t - E;
    int d = t < E ? dst[t] : t - E;
    #pragma unroll
    for (int h = 0; h < HEADS; ++h) {
        float e = es[s * HEADS + h] + ed[d * HEADS + h];
        e = e > 0.f ? e : NEG_SLOPE * e;
        float m = ord2f(mu[d * HEADS + h]);
        atomicAdd(&denom[d * HEADS + h], __expf(e - m));
    }
}

// Pass 3: acc[dst,:] += Wh[src,:] * alpha.  One wave per edge; lane owns 4 dims
// (head = lane>>2). alpha recomputed from es/ed/mu/denom (no per-edge storage).
__global__ void __launch_bounds__(256)
edge_accum_kernel(const int* __restrict__ src, const int* __restrict__ dst,
                  int E, int N,
                  const float* __restrict__ es, const float* __restrict__ ed,
                  const unsigned* __restrict__ mu, const float* __restrict__ denom,
                  const float* __restrict__ Wh, float* __restrict__ acc) {
    long long gid  = (long long)blockIdx.x * blockDim.x + threadIdx.x;
    long long edge = gid >> 5;
    int       lane = (int)(gid & 31);
    if (edge >= (long long)E + N) return;
    int s = edge < E ? src[edge] : (int)(edge - E);
    int d = edge < E ? dst[edge] : (int)(edge - E);
    int hh = lane >> 2;
    float e = es[s * HEADS + hh] + ed[d * HEADS + hh];
    e = e > 0.f ? e : NEG_SLOPE * e;
    float m     = ord2f(mu[d * HEADS + hh]);
    float alpha = __expf(e - m) / denom[d * HEADS + hh];
    const float4 w = *(const float4*)(Wh + (size_t)s * DIMK + lane * 4);
    float* o = acc + (size_t)d * DIMK + lane * 4;
    atomicAdd(o + 0, w.x * alpha);
    atomicAdd(o + 1, w.y * alpha);
    atomicAdd(o + 2, w.z * alpha);
    atomicAdd(o + 3, w.w * alpha);
}

__global__ void elu_kernel(float* __restrict__ x, long long n) {
    long long t = (long long)blockIdx.x * blockDim.x + threadIdx.x;
    if (t >= n) return;
    float v = x[t];
    x[t] = v > 0.f ? v : (__expf(v) - 1.f);
}

// align[i,:] = out_sr[sr_data[i],:] - out_tg[tg_data[i],:]   (float4 per thread)
__global__ void align_kernel(const float* __restrict__ outS, const float* __restrict__ outT,
                             const int* __restrict__ srd, const int* __restrict__ tgd,
                             float* __restrict__ out, int A) {
    long long t = (long long)blockIdx.x * blockDim.x + threadIdx.x;
    if (t >= (long long)A * (DIMK / 4)) return;
    int i = (int)(t >> 5), c = ((int)t & 31) * 4;
    float4 a = *(const float4*)(outS + (size_t)srd[i] * DIMK + c);
    float4 b = *(const float4*)(outT + (size_t)tgd[i] * DIMK + c);
    float4 r = { a.x - b.x, a.y - b.y, a.z - b.z, a.w - b.w };
    *(float4*)(out + (size_t)i * DIMK + c) = r;
}

// transe[i,:] = ent[h[i],:] + rel[r[i],:] - ent[t[i],:]
__global__ void transe_kernel(const float* __restrict__ ent, const float* __restrict__ rel,
                              const int* __restrict__ hl, const int* __restrict__ tl,
                              const int* __restrict__ rl,
                              float* __restrict__ out, int T) {
    long long t = (long long)blockIdx.x * blockDim.x + threadIdx.x;
    if (t >= (long long)T * (DIMK / 4)) return;
    int i = (int)(t >> 5), c = ((int)t & 31) * 4;
    float4 a = *(const float4*)(ent + (size_t)hl[i] * DIMK + c);
    float4 r = *(const float4*)(rel + (size_t)rl[i] * DIMK + c);
    float4 b = *(const float4*)(ent + (size_t)tl[i] * DIMK + c);
    float4 o = { a.x + r.x - b.x, a.y + r.y - b.y, a.z + r.z - b.z, a.w + r.w - b.w };
    *(float4*)(out + (size_t)i * DIMK + c) = o;
}

// ---------------------------------------------------------------------------

extern "C" void kernel_launch(void* const* d_in, const int* in_sizes, int n_in,
                              void* d_out, int out_size, void* d_ws, size_t ws_size,
                              hipStream_t stream) {
    const float* emb_sr = (const float*)d_in[0];
    const float* emb_tg = (const float*)d_in[1];
    const float* rel_sr = (const float*)d_in[2];
    const float* rel_tg = (const float*)d_in[3];
    const float* W      = (const float*)d_in[4];
    const float* a_src  = (const float*)d_in[5];
    const float* a_dst  = (const float*)d_in[6];
    const int* src_sr  = (const int*)d_in[7];
    const int* dst_sr  = (const int*)d_in[8];
    const int* src_tg  = (const int*)d_in[9];
    const int* dst_tg  = (const int*)d_in[10];
    const int* sr_data = (const int*)d_in[11];
    const int* tg_data = (const int*)d_in[12];
    const int* h_sr = (const int*)d_in[13];
    const int* t_sr = (const int*)d_in[14];
    const int* r_sr = (const int*)d_in[15];
    const int* h_tg = (const int*)d_in[16];
    const int* t_tg = (const int*)d_in[17];
    const int* r_tg = (const int*)d_in[18];

    const int Nsr = in_sizes[0] / DIMK;
    const int Ntg = in_sizes[1] / DIMK;
    const int Esr = in_sizes[7];
    const int Etg = in_sizes[9];
    const int A   = in_sizes[11];
    const int Tsr = in_sizes[13];
    const int Ttg = in_sizes[16];
    const int L   = in_sizes[4] / (HEADS * DIMK * DHEAD);
    const int maxN = Nsr > Ntg ? Nsr : Ntg;

    // ---- carve workspace (256B aligned chunks) ----
    char* p = (char*)d_ws;
    auto carve = [&](size_t bytes) {
        char* q = p;
        p += (bytes + 255) & ~(size_t)255;
        return (void*)q;
    };
    float*    Wh    = (float*)carve((size_t)maxN * DIMK * sizeof(float));
    float*    hbuf0 = (float*)carve((size_t)maxN * DIMK * sizeof(float));
    float*    hbuf1 = (L > 2) ? (float*)carve((size_t)maxN * DIMK * sizeof(float)) : hbuf0;
    float*    outS  = (float*)carve((size_t)Nsr * DIMK * sizeof(float));
    float*    outT  = (float*)carve((size_t)Ntg * DIMK * sizeof(float));
    float*    es    = (float*)carve((size_t)maxN * HEADS * sizeof(float));
    float*    ed    = (float*)carve((size_t)maxN * HEADS * sizeof(float));
    unsigned* mu    = (unsigned*)carve((size_t)maxN * HEADS * sizeof(unsigned));
    float*    denom = (float*)carve((size_t)maxN * HEADS * sizeof(float));
    __bf16*   gB    = (__bf16*)carve((size_t)4 * 8 * 32 * 16 * sizeof(__bf16));

    float* hping[2] = { hbuf0, hbuf1 };

    for (int g = 0; g < 2; ++g) {
        const int    N    = g == 0 ? Nsr : Ntg;
        const int    E    = g == 0 ? Esr : Etg;
        const int*   srcp = g == 0 ? src_sr : src_tg;
        const int*   dstp = g == 0 ? dst_sr : dst_tg;
        const float* h0   = g == 0 ? emb_sr : emb_tg;
        float*       outg = g == 0 ? outS : outT;

        for (int l = 0; l < L; ++l) {
            const float* hin = (l == 0) ? h0 : hping[(l - 1) & 1];
            float*       acc = (l == L - 1) ? outg : hping[l & 1];

            prep_B_kernel<<<1, 256, 0, stream>>>(W, l, gB);

            gat_gemm_wmma<<<(N + 127) / 128, 256, 0, stream>>>(hin, gB, Wh, N);

            const int nh = N * HEADS;
            attn_coef_kernel<<<(nh + 255) / 256, 256, 0, stream>>>(
                Wh, a_src + (size_t)l * HEADS * DHEAD, a_dst + (size_t)l * HEADS * DHEAD,
                es, ed, N);

            const long long n1 = (long long)N * DIMK;
            zero_kernel<<<(unsigned)((n1 + 255) / 256), 256, 0, stream>>>(
                acc, n1, mu, denom, (long long)nh);

            const int Etot = E + N;   // real edges + self-loops
            edge_max_kernel<<<(Etot + 255) / 256, 256, 0, stream>>>(
                srcp, dstp, E, N, es, ed, mu);
            edge_sum_kernel<<<(Etot + 255) / 256, 256, 0, stream>>>(
                srcp, dstp, E, N, es, ed, mu, denom);

            const long long accT = (long long)Etot * 32;
            edge_accum_kernel<<<(unsigned)((accT + 255) / 256), 256, 0, stream>>>(
                srcp, dstp, E, N, es, ed, mu, denom, Wh, acc);

            elu_kernel<<<(unsigned)((n1 + 255) / 256), 256, 0, stream>>>(acc, n1);
        }
    }

    // ---- outputs: [align | sr_transe | tg_transe], all f32, DIM=128 wide ----
    float* out = (float*)d_out;
    const long long aT = (long long)A * (DIMK / 4);
    align_kernel<<<(unsigned)((aT + 255) / 256), 256, 0, stream>>>(
        outS, outT, sr_data, tg_data, out, A);

    float* out1 = out + (size_t)A * DIMK;
    const long long t1 = (long long)Tsr * (DIMK / 4);
    transe_kernel<<<(unsigned)((t1 + 255) / 256), 256, 0, stream>>>(
        outS, rel_sr, h_sr, t_sr, r_sr, out1, Tsr);

    float* out2 = out1 + (size_t)Tsr * DIMK;
    const long long t2 = (long long)Ttg * (DIMK / 4);
    transe_kernel<<<(unsigned)((t2 + 255) / 256), 256, 0, stream>>>(
        outT, rel_tg, h_tg, t_tg, r_tg, out2, Ttg);
}
